// Attn_15144054686375
// MI455X (gfx1250) — compile-verified, matrix operand order checked by reading, not run on hardware
//
#include <hip/hip_runtime.h>

// ---------------------------------------------------------------------------
// Problem constants
// ---------------------------------------------------------------------------
#define BATCH   4
#define SEQ     2048
#define NHEAD   16
#define NEMBD   2048
#define HDIM    128                       // NEMBD / NHEAD
#define NOUT    6144                      // 3 * NEMBD
#define YSIZE   ((size_t)BATCH * SEQ * NEMBD)   // floats per output tensor
#define QKVSZ   ((size_t)BATCH * NHEAD * SEQ * HDIM) // == YSIZE elements

typedef __attribute__((ext_vector_type(4)))  __bf16 v4bf;
typedef __attribute__((ext_vector_type(8)))  __bf16 v8bf;
typedef __attribute__((ext_vector_type(16))) __bf16 v16bf;
typedef __attribute__((ext_vector_type(8)))  float  v8f;
typedef __attribute__((ext_vector_type(4)))  int    v4i;

// fp32 -> bf16, round-to-nearest-even
__device__ __forceinline__ __bf16 f2bf(float f) {
    unsigned int u = __builtin_bit_cast(unsigned int, f);
    unsigned int r = (u + 0x7FFFu + ((u >> 16) & 1u)) >> 16;
    unsigned short s = (unsigned short)r;
    return __builtin_bit_cast(__bf16, s);
}

// ---------------------------------------------------------------------------
// CDNA5 async global->LDS copy (16 bytes per lane), with fallback.
// Builtin signature (from hipcc diagnostic): param0 = int4 addrspace(1)*,
// param1 = int4 addrspace(3)*, plus two int immediates (offset, cpol).
// ---------------------------------------------------------------------------
#if defined(__AMDGCN__) && __has_builtin(__builtin_amdgcn_global_load_async_to_lds_b128)
#define ASYNC_LDS 1
#else
#define ASYNC_LDS 0
#endif

#if ASYNC_LDS
typedef __attribute__((address_space(1))) v4i as1_v4i;
typedef __attribute__((address_space(3))) v4i as3_v4i;
#endif

__device__ __forceinline__ void async_copy16(const __bf16* g, __bf16* l) {
#if ASYNC_LDS
    __builtin_amdgcn_global_load_async_to_lds_b128(
        (as1_v4i*)g, (as3_v4i*)l, 0, 0);
#else
    *(uint4*)l = *(const uint4*)g;
#endif
}

__device__ __forceinline__ void wait_async() {
#if ASYNC_LDS
#if __has_builtin(__builtin_amdgcn_s_wait_asynccnt)
    __builtin_amdgcn_s_wait_asynccnt(0);
#else
    asm volatile("s_wait_asynccnt 0x0" ::: "memory");
#endif
#endif
}

// ---------------------------------------------------------------------------
// WMMA bf16 fragment load from an LDS tile (row-major, `stride` elems/row).
// ISA layout (A 16x32 / B 32x16, symmetric):
//   lanes 0-15 : row/col = lane,    K = kofs+[0..7] and kofs+16+[0..7]
//   lanes 16-31: row/col = lane-16, K = kofs+8+[0..7] and kofs+24+[0..7]
// ---------------------------------------------------------------------------
__device__ __forceinline__ v16bf load_frag(const __bf16* tile, int stride,
                                           int kofs, int lane) {
    const int lr = lane & 15;
    const int h2 = lane >> 4;
    const __bf16* p = tile + lr * stride + kofs + h2 * 8;
    union { v16bf v; v8bf h[2]; } u;
    u.h[0] = *(const v8bf*)(p);
    u.h[1] = *(const v8bf*)(p + 16);
    return u.v;
}

// 16-lane-half reductions (masks 1..8 stay within each wave32 half)
__device__ __forceinline__ float half_max(float v) {
    v = fmaxf(v, __shfl_xor(v, 1, 32));
    v = fmaxf(v, __shfl_xor(v, 2, 32));
    v = fmaxf(v, __shfl_xor(v, 4, 32));
    v = fmaxf(v, __shfl_xor(v, 8, 32));
    return v;
}
__device__ __forceinline__ float half_sum(float v) {
    v += __shfl_xor(v, 1, 32);
    v += __shfl_xor(v, 2, 32);
    v += __shfl_xor(v, 4, 32);
    v += __shfl_xor(v, 8, 32);
    return v;
}

// ---------------------------------------------------------------------------
// Kernel 0: fp32 -> bf16 bulk conversion (x and W done once, then the GEMM
// streams half the bytes and stages with zero VALU).
// ---------------------------------------------------------------------------
__global__ __launch_bounds__(256, 1)
void cvt_bf16_kernel(const float* __restrict__ in, __bf16* __restrict__ out,
                     int n4) {
    const int i = blockIdx.x * 256 + threadIdx.x;
    if (i < n4) {
        const float4 f = ((const float4*)in)[i];
        v4bf o;
        o[0] = f2bf(f.x); o[1] = f2bf(f.y); o[2] = f2bf(f.z); o[3] = f2bf(f.w);
        ((v4bf*)out)[i] = o;
    }
}

// ---------------------------------------------------------------------------
// Kernel 1: QKV projection  C[8192,6144] = x @ W^T + b.
// Block tile 128x128, K-step 64, double-buffered async LDS staging.
// Epilogue scatters: q -> qb (bf16, pre-scaled by 1/sqrt(D), [bh][s][d])
//                    k -> d_out fp32 [bh][s][d] + kb bf16 [bh][s][d]
//                    v -> d_out fp32 [bh][s][d] + vtb bf16 [bh][d][s]
// ---------------------------------------------------------------------------
__global__ __launch_bounds__(256, 1)
void qkv_gemm_kernel(const __bf16* __restrict__ xb, const __bf16* __restrict__ Wb,
                     const float* __restrict__ bias,
                     __bf16* __restrict__ qb, __bf16* __restrict__ kb,
                     __bf16* __restrict__ vtb,
                     float* __restrict__ ko, float* __restrict__ vo) {
    __shared__ __bf16 Asm[2][128][72];   // 64 cols + 8 pad (row = 144B, 16B mult)
    __shared__ __bf16 Bsm[2][128][72];

    const int tid  = threadIdx.x;
    const int lane = tid & 31;
    const int w    = tid >> 5;        // 0..7
    const int wm   = w >> 2;          // 0..1 -> 64 rows each
    const int wn   = w & 3;           // 0..3 -> 32 cols each
    const int m0   = blockIdx.y * 128;
    const int n0   = blockIdx.x * 128;
    const int half = lane >> 4;
    const int lc   = lane & 15;

    v8f acc[4][2];
#pragma unroll
    for (int i = 0; i < 4; ++i)
#pragma unroll
        for (int j = 0; j < 2; ++j)
#pragma unroll
            for (int e = 0; e < 8; ++e) acc[i][j][e] = 0.0f;

    // stage one 128x64 bf16 tile pair: 1024 16B chunks each, 4 per thread
    auto stage = [&](int buf, int k0) {
#pragma unroll
        for (int i = 0; i < 4; ++i) {
            const int idx = tid + i * 256;
            const int r   = idx >> 3;
            const int c8  = idx & 7;
            async_copy16(xb + (size_t)(m0 + r) * NEMBD + k0 + c8 * 8,
                         &Asm[buf][r][c8 * 8]);
            async_copy16(Wb + (size_t)(n0 + r) * NEMBD + k0 + c8 * 8,
                         &Bsm[buf][r][c8 * 8]);
        }
    };

    stage(0, 0);
    wait_async();
    __syncthreads();

    for (int k0 = 0; k0 < NEMBD; k0 += 64) {
        const int buf = (k0 >> 6) & 1;
        if (k0 + 64 < NEMBD) stage(buf ^ 1, k0 + 64);   // overlap with compute

#pragma unroll
        for (int kc = 0; kc < 2; ++kc) {
            v16bf afr[4], bfr[2];
#pragma unroll
            for (int i = 0; i < 4; ++i)
                afr[i] = load_frag(&Asm[buf][wm * 64 + i * 16][0], 72, kc * 32, lane);
#pragma unroll
            for (int j = 0; j < 2; ++j)
                bfr[j] = load_frag(&Bsm[buf][wn * 32 + j * 16][0], 72, kc * 32, lane);
#pragma unroll
            for (int i = 0; i < 4; ++i)
#pragma unroll
                for (int j = 0; j < 2; ++j)
                    acc[i][j] = __builtin_amdgcn_wmma_f32_16x16x32_bf16(
                        false, afr[i], false, bfr[j], (short)0, acc[i][j],
                        false, false);
        }
        wait_async();
        __syncthreads();
    }

    // ---- epilogue ----
    const float qscale = 0.08838834764831845f;   // 1/sqrt(128)
#pragma unroll
    for (int i = 0; i < 4; ++i) {
#pragma unroll
        for (int j = 0; j < 2; ++j) {
            const int  nb = n0 + wn * 32 + j * 16 + lc;   // output column
            const int  p  = nb >> 11;                     // 0=q 1=k 2=v
            const int  e  = nb & 2047;
            const int  h  = e >> 7;
            const int  d  = e & 127;
            const float bv = bias[nb];
#pragma unroll
            for (int r = 0; r < 8; ++r) {
                const int mrow = m0 + wm * 64 + i * 16 + r + half * 8;
                const int b    = mrow >> 11;              // SEQ = 2048
                const int s    = mrow & 2047;
                const size_t bh = (size_t)b * NHEAD + h;
                const size_t oi = (bh * SEQ + s) * HDIM + d;
                const float val = acc[i][j][r] + bv;
                if (p == 0) {
                    qb[oi] = f2bf(val * qscale);
                } else if (p == 1) {
                    ko[oi] = val;
                    kb[oi] = f2bf(val);
                } else {
                    vo[oi] = val;
                    vtb[(bh * HDIM + d) * SEQ + s] = f2bf(val);   // transposed
                }
            }
        }
    }
}

// ---------------------------------------------------------------------------
// Kernel 2: flash-attention (non-causal). One block = one (b,h), 128 queries,
// 8 waves x 16 queries. KV tiled by 64, double-buffered async staging.
// ---------------------------------------------------------------------------
__global__ __launch_bounds__(256, 1)
void attn_kernel(const __bf16* __restrict__ qb, const __bf16* __restrict__ kb,
                 const __bf16* __restrict__ vtb, float* __restrict__ y) {
    __shared__ __bf16 Ksm[2][64][136];   // [kv][d]  row = 272B (16B mult)
    __shared__ __bf16 Vts[2][128][72];   // [d][kv]  row = 144B (16B mult)
    __shared__ __bf16 Psm[8][16][72];    // per-wave P, [qrow][kv]

    const int tid  = threadIdx.x;
    const int lane = tid & 31;
    const int w    = tid >> 5;
    const int half = lane >> 4;
    const int lc   = lane & 15;
    const int bh   = blockIdx.y;               // b*16 + h
    const int q0   = blockIdx.x * 128;         // query base
    const size_t base = (size_t)bh * SEQ * HDIM; // same stride for [s][d] & [d][s]

    // ---- Q fragments straight from global bf16 (pre-scaled at epilogue) ----
    v16bf qf[4];
    {
        const __bf16* qp = qb + base + (size_t)(q0 + w * 16 + lc) * HDIM;
#pragma unroll
        for (int kc = 0; kc < 4; ++kc) {
            union { v16bf vv; v8bf h[2]; } u;
            u.h[0] = *(const v8bf*)(qp + kc * 32 + half * 8);
            u.h[1] = *(const v8bf*)(qp + kc * 32 + 16 + half * 8);
            qf[kc] = u.vv;
        }
    }

    float mrow[8], lrow[8];
    v8f Oa[8];
#pragma unroll
    for (int r = 0; r < 8; ++r) { mrow[r] = -1e30f; lrow[r] = 0.0f; }
#pragma unroll
    for (int t = 0; t < 8; ++t)
#pragma unroll
        for (int e = 0; e < 8; ++e) Oa[t][e] = 0.0f;

    // stage a 64-key tile: K 64x128 and Vt 128x64, 1024 16B chunks each
    auto stage = [&](int buf, int kv0) {
#pragma unroll
        for (int i = 0; i < 4; ++i) {
            const int idx = tid + i * 256;
            {
                const int r  = idx >> 4;        // kv row 0..63
                const int c8 = idx & 15;        // d chunk
                async_copy16(kb + base + (size_t)(kv0 + r) * HDIM + c8 * 8,
                             &Ksm[buf][r][c8 * 8]);
            }
            {
                const int r  = idx >> 3;        // d row 0..127
                const int c8 = idx & 7;         // kv chunk
                async_copy16(vtb + base + (size_t)r * SEQ + kv0 + c8 * 8,
                             &Vts[buf][r][c8 * 8]);
            }
        }
    };

    stage(0, 0);
    wait_async();
    __syncthreads();

    for (int kv0 = 0; kv0 < SEQ; kv0 += 64) {
        const int buf = (kv0 >> 6) & 1;
        if (kv0 + 64 < SEQ) stage(buf ^ 1, kv0 + 64);   // overlap with compute

        // ---- S = Q * K^T (16 queries x 64 keys), contraction over d=128 ----
        v8f S[4];
#pragma unroll
        for (int j = 0; j < 4; ++j) {
#pragma unroll
            for (int e = 0; e < 8; ++e) S[j][e] = 0.0f;
#pragma unroll
            for (int kc = 0; kc < 4; ++kc) {
                const v16bf bf = load_frag(&Ksm[buf][j * 16][0], 136, kc * 32, lane);
                S[j] = __builtin_amdgcn_wmma_f32_16x16x32_bf16(
                    false, qf[kc], false, bf, (short)0, S[j], false, false);
            }
        }

        // ---- online softmax update ----
        float nm[8], sf[8], rs[8];
#pragma unroll
        for (int r = 0; r < 8; ++r) {
            float vmx = fmaxf(fmaxf(S[0][r], S[1][r]), fmaxf(S[2][r], S[3][r]));
            vmx   = half_max(vmx);
            nm[r] = fmaxf(mrow[r], vmx);
            sf[r] = __expf(mrow[r] - nm[r]);
            mrow[r] = nm[r];
            rs[r] = 0.0f;
        }
#pragma unroll
        for (int j = 0; j < 4; ++j)
#pragma unroll
            for (int r = 0; r < 8; ++r) {
                const float p = __expf(S[j][r] - nm[r]);
                S[j][r] = p;
                rs[r] += p;
            }
#pragma unroll
        for (int r = 0; r < 8; ++r) {
            lrow[r] = lrow[r] * sf[r] + half_sum(rs[r]);
#pragma unroll
            for (int t = 0; t < 8; ++t) Oa[t][r] *= sf[r];
        }

        // ---- C-layout P -> A-fragment layout via per-wave LDS ----
#pragma unroll
        for (int j = 0; j < 4; ++j)
#pragma unroll
            for (int r = 0; r < 8; ++r)
                Psm[w][r + half * 8][j * 16 + lc] = f2bf(S[j][r]);

        // ---- O += P * V  (contraction over 64 keys) ----
#pragma unroll
        for (int kc = 0; kc < 2; ++kc) {
            const v16bf pf = load_frag(&Psm[w][0][0], 72, kc * 32, lane);
#pragma unroll
            for (int t = 0; t < 8; ++t) {
                const v16bf vf = load_frag(&Vts[buf][t * 16][0], 72, kc * 32, lane);
                Oa[t] = __builtin_amdgcn_wmma_f32_16x16x32_bf16(
                    false, pf, false, vf, (short)0, Oa[t], false, false);
            }
        }

        wait_async();      // next tile landed
        __syncthreads();   // everyone done reading this tile
    }

    // ---- normalize and write y back to [B*S, NEMBD] ----
    const int b = bh >> 4;
    const int h = bh & 15;
#pragma unroll
    for (int r = 0; r < 8; ++r) {
        const int   row = q0 + w * 16 + r + half * 8;
        const float inv = 1.0f / lrow[r];
        float* yp = y + (size_t)(b * SEQ + row) * NEMBD + h * HDIM + lc;
#pragma unroll
        for (int t = 0; t < 8; ++t)
            yp[t * 16] = Oa[t][r] * inv;
    }
}

// ---------------------------------------------------------------------------
// Launch
// ---------------------------------------------------------------------------
extern "C" void kernel_launch(void* const* d_in, const int* in_sizes, int n_in,
                              void* d_out, int out_size, void* d_ws, size_t ws_size,
                              hipStream_t stream) {
    const float* x = (const float*)d_in[0];
    const float* W = (const float*)d_in[1];
    const float* b = (const float*)d_in[2];

    float* y  = (float*)d_out;                 // [B*S, NEMBD]
    float* ko = y + YSIZE;                     // [B,H,S,D] fp32 output
    float* vo = y + 2 * YSIZE;                 // [B,H,S,D] fp32 output

    // workspace layout (bf16): xb | Wb | qb | kb | vtb  = 152 MB total
    __bf16* xb  = (__bf16*)d_ws;
    __bf16* Wb  = xb + (size_t)(BATCH * SEQ) * NEMBD;
    __bf16* qb  = Wb + (size_t)NOUT * NEMBD;
    __bf16* kb  = qb + QKVSZ;
    __bf16* vtb = kb + QKVSZ;

    // 0) one-time fp32 -> bf16 conversion of GEMM operands
    const int xn4 = (int)((size_t)(BATCH * SEQ) * NEMBD / 4);
    const int wn4 = (int)((size_t)NOUT * NEMBD / 4);
    cvt_bf16_kernel<<<(xn4 + 255) / 256, 256, 0, stream>>>(x, xb, xn4);
    cvt_bf16_kernel<<<(wn4 + 255) / 256, 256, 0, stream>>>(W, Wb, wn4);

    // 1) QKV projection: 48 N-tiles x 64 M-tiles
    dim3 g1(NOUT / 128, (BATCH * SEQ) / 128);
    qkv_gemm_kernel<<<g1, 256, 0, stream>>>(xb, Wb, b, qb, kb, vtb, ko, vo);

    // 2) Attention: 16 query chunks x 64 (b,h) pairs
    dim3 g2(SEQ / 128, BATCH * NHEAD);
    attn_kernel<<<g2, 256, 0, stream>>>(qb, kb, vtb, y);
}
